// MLPDecoder_34832184770749
// MI455X (gfx1250) — matrix operand order repaired and checked by reading, not compiled, not run on hardware
//
#include <hip/hip_runtime.h>
#include <hip/hip_bf16.h>

// ---------------------------------------------------------------------------
// MI455X / gfx1250 implementation.
//   K1: per-graph MLPs (number_nodes head + latent transform e[B,16])
//   K2: GIN (fully-connected graphs => msg = per-graph column sum), LN+leaky,
//       node-feature MLP, batch passthrough. fp32 VALU, wave32 butterflies.
//   K3: edge-type MLP on v_wmma_f32_16x16x32_f16 over 16-pair tiles,
//       symmetric-pair dedup (496 pairs/graph -> writes both edge slots),
//       edge_index passthrough, global_prefetch_b8 for next tile.
// d_in leaf order (recursive insertion-order flatten of setup_inputs()):
//  0 latent_vec  1 batch  2 edge_index  3 noise
//  4..9  node_num (W,b)x3      10..12 latent W x3
//  13+6l..18+6l conv l: W1,b1,g1,bt1,W2,b2 (l=0..3)
//  37..42 norms g,b x3         43..48 node_feat (W,b)x3
//  49..54 edge_type (W,b)x3
// ---------------------------------------------------------------------------

typedef __attribute__((ext_vector_type(16))) _Float16 v16h;
typedef __attribute__((ext_vector_type(8)))  float    v8f;

#define WMMA_F16(A,B,C) \
  __builtin_amdgcn_wmma_f32_16x16x32_f16(false,(A),false,(B),(short)0,(C),false,false)

constexpr int    BG   = 2048;
constexpr int    NPG  = 32;
constexpr int    NN   = BG * NPG;          // 65536
constexpr int    EPG  = 992;
constexpr long   ETOT = (long)BG * EPG;    // 2031616
constexpr size_t OUT_NF    = 0;
constexpr size_t OUT_EI    = (size_t)NN * 2;              // 131072
constexpr size_t OUT_ATTR  = OUT_EI + 2 * (size_t)ETOT;   // 4194304
constexpr size_t OUT_BATCH = OUT_ATTR + (size_t)ETOT * 5; // 14352384
constexpr size_t OUT_NNUM  = OUT_BATCH + NN;              // 14417920

// ============================ K1: graph MLPs ===============================
__global__ __launch_bounds__(64) void k_graph(
    const float* __restrict__ lat,
    const float* __restrict__ Wn1, const float* __restrict__ bn1,
    const float* __restrict__ Wn2, const float* __restrict__ bn2,
    const float* __restrict__ Wn3, const float* __restrict__ bn3,
    const float* __restrict__ Wl1, const float* __restrict__ Wl2,
    const float* __restrict__ Wl3,
    float* __restrict__ out_nn, float* __restrict__ ws_e)
{
  __shared__ float sl[128];
  __shared__ float t1[64];
  __shared__ float t2[32];
  const int g = blockIdx.x, tid = threadIdx.x;
  sl[tid]      = lat[g * 128 + tid];
  sl[tid + 64] = lat[g * 128 + 64 + tid];
  __syncthreads();
  // number_nodes head
  { float a = bn1[tid];
    for (int k = 0; k < 128; ++k) a += sl[k] * Wn1[k * 64 + tid];
    t1[tid] = fmaxf(a, 0.f); }
  __syncthreads();
  if (tid < 32) { float a = bn2[tid];
    for (int k = 0; k < 64; ++k) a += t1[k] * Wn2[k * 32 + tid];
    t2[tid] = fmaxf(a, 0.f); }
  __syncthreads();
  if (tid == 0) { float a = bn3[0];
    for (int k = 0; k < 32; ++k) a += t2[k] * Wn3[k];
    out_nn[g] = a; }
  __syncthreads();
  // latent transform (bias-free, plain last layer)
  { float a = 0.f;
    for (int k = 0; k < 128; ++k) a += sl[k] * Wl1[k * 64 + tid];
    t1[tid] = fmaxf(a, 0.f); }
  __syncthreads();
  if (tid < 32) { float a = 0.f;
    for (int k = 0; k < 64; ++k) a += t1[k] * Wl2[k * 32 + tid];
    t2[tid] = fmaxf(a, 0.f); }
  __syncthreads();
  if (tid < 16) { float a = 0.f;
    for (int k = 0; k < 32; ++k) a += t2[k] * Wl3[k * 16 + tid];
    ws_e[g * 16 + tid] = a; }
}

// ============================ K2: GIN + node MLP ===========================
struct GnnArgs {
  const float* W1[4]; const float* b1[4]; const float* g1[4]; const float* bt1[4];
  const float* W2[4]; const float* b2[4];
  const float* ng[3]; const float* nb[3];
  const float* Wf1; const float* bf1; const float* Wf2; const float* bf2;
  const float* Wf3; const float* bf3;
};

__device__ __forceinline__ void cpLds(float* dst, const float* src, int n,
                                      int tid, int nt) {
  for (int i = tid; i < n; i += nt) dst[i] = src[i];
}

__global__ __launch_bounds__(256) void k_gnn(
    GnnArgs P, const float* __restrict__ ws_e, const float* __restrict__ noise,
    const int* __restrict__ batch, float* __restrict__ ws_x,
    float* __restrict__ out)
{
  __shared__ float S[4098];
  constexpr int oW1[4] = {0, 288, 544, 800};
  constexpr int oB1 = 1056, oG1 = 1120, oBT = 1184;
  constexpr int oW2 = 1248, oB2 = 2272;
  constexpr int oNG = 2336, oNB = 2384;
  constexpr int oF1 = 2432, oFB1 = 2944, oF2 = 2976, oFB2 = 4000,
                oF3 = 4032, oFB3 = 4096;
  const int tid = threadIdx.x;
  for (int l = 0; l < 4; ++l) {
    const int kd = (l == 0) ? 18 : 16;
    cpLds(S + oW1[l],      P.W1[l],  kd * 16, tid, 256);
    cpLds(S + oB1 + l*16,  P.b1[l],  16, tid, 256);
    cpLds(S + oG1 + l*16,  P.g1[l],  16, tid, 256);
    cpLds(S + oBT + l*16,  P.bt1[l], 16, tid, 256);
    cpLds(S + oW2 + l*256, P.W2[l],  256, tid, 256);
    cpLds(S + oB2 + l*16,  P.b2[l],  16, tid, 256);
  }
  for (int l = 0; l < 3; ++l) {
    cpLds(S + oNG + l*16, P.ng[l], 16, tid, 256);
    cpLds(S + oNB + l*16, P.nb[l], 16, tid, 256);
  }
  cpLds(S + oF1,  P.Wf1, 512, tid, 256);  cpLds(S + oFB1, P.bf1, 32, tid, 256);
  cpLds(S + oF2,  P.Wf2, 1024, tid, 256); cpLds(S + oFB2, P.bf2, 32, tid, 256);
  cpLds(S + oF3,  P.Wf3, 64, tid, 256);   cpLds(S + oFB3, P.bf3, 2, tid, 256);
  __syncthreads();

  const int lane = tid & 31, wid = tid >> 5;
  const int g = blockIdx.x * 8 + wid;      // 256 blocks * 8 waves == 2048
  const int n = g * NPG + lane;            // one wave32 == one graph

  float x[18];
  #pragma unroll
  for (int f = 0; f < 16; ++f) x[f] = ws_e[g * 16 + f];
  x[16] = noise[2 * n]; x[17] = noise[2 * n + 1];

  int K = 18;
  for (int l = 0; l < 4; ++l) {
    // fully connected + self loop  =>  msg_i = sum over all 32 nodes
    float h[18];
    for (int f = 0; f < K; ++f) {
      float s = x[f];
      s += __shfl_xor(s, 16, 32); s += __shfl_xor(s, 8, 32);
      s += __shfl_xor(s, 4, 32);  s += __shfl_xor(s, 2, 32);
      s += __shfl_xor(s, 1, 32);
      h[f] = x[f] + s;
    }
    const float* W1 = S + oW1[l];
    float z[16];
    for (int o = 0; o < 16; ++o) {
      float a = S[oB1 + l*16 + o];
      for (int k = 0; k < K; ++k) a += h[k] * W1[k * 16 + o];
      z[o] = a;
    }
    float mu = 0.f;
    #pragma unroll
    for (int o = 0; o < 16; ++o) mu += z[o];
    mu *= (1.f / 16.f);
    float var = 0.f;
    #pragma unroll
    for (int o = 0; o < 16; ++o) { float dv = z[o] - mu; var += dv * dv; }
    var *= (1.f / 16.f);
    float rin = rsqrtf(var + 1e-5f);
    #pragma unroll
    for (int o = 0; o < 16; ++o) {
      float v = (z[o] - mu) * rin * S[oG1 + l*16 + o] + S[oBT + l*16 + o];
      z[o] = (v > 0.f) ? v : 0.01f * v;
    }
    const float* W2 = S + oW2 + l * 256;
    for (int o = 0; o < 16; ++o) {
      float a = S[oB2 + l*16 + o];
      #pragma unroll
      for (int k = 0; k < 16; ++k) a += z[k] * W2[k * 16 + o];
      x[o] = a;
    }
    K = 16;
    if (l < 3) {
      float m2 = 0.f;
      #pragma unroll
      for (int o = 0; o < 16; ++o) m2 += x[o];
      m2 *= (1.f / 16.f);
      float v2 = 0.f;
      #pragma unroll
      for (int o = 0; o < 16; ++o) { float dv = x[o] - m2; v2 += dv * dv; }
      v2 *= (1.f / 16.f);
      float r2 = rsqrtf(v2 + 1e-5f);
      #pragma unroll
      for (int o = 0; o < 16; ++o) {
        float v = (x[o] - m2) * r2 * S[oNG + l*16 + o] + S[oNB + l*16 + o];
        x[o] = (v > 0.f) ? v : 0.01f * v;
      }
    }
  }
  #pragma unroll
  for (int f = 0; f < 16; ++f) ws_x[(size_t)n * 16 + f] = x[f];
  // node-feature MLP 16->32->32->2
  float t1[32];
  for (int o = 0; o < 32; ++o) {
    float a = S[oFB1 + o];
    #pragma unroll
    for (int k = 0; k < 16; ++k) a += x[k] * S[oF1 + k * 32 + o];
    t1[o] = fmaxf(a, 0.f);
  }
  float t2[32];
  for (int o = 0; o < 32; ++o) {
    float a = S[oFB2 + o];
    #pragma unroll
    for (int k = 0; k < 32; ++k) a += t1[k] * S[oF2 + k * 32 + o];
    t2[o] = fmaxf(a, 0.f);
  }
  #pragma unroll
  for (int c = 0; c < 2; ++c) {
    float a = S[oFB3 + c];
    #pragma unroll
    for (int k = 0; k < 32; ++k) a += t2[k] * S[oF3 + k * 2 + c];
    out[OUT_NF + (size_t)n * 2 + c] = a;
  }
  out[OUT_BATCH + n] = (float)batch[n];
}

// ============================ K3: edge MLP (WMMA) ==========================
__global__ __launch_bounds__(256) void k_edge(
    const float* __restrict__ ws_x, const int* __restrict__ edge_index,
    const float* __restrict__ We1, const float* __restrict__ be1,
    const float* __restrict__ We2, const float* __restrict__ be2,
    const float* __restrict__ We3, const float* __restrict__ be3,
    float* __restrict__ out)
{
  __shared__ _Float16 sWm[32 * 64];     // folded [m,m] layer-1 weight, K-padded
  __shared__ _Float16 sW2[64 * 64];
  __shared__ _Float16 sW3[64 * 16];     // N-padded 5 -> 16
  __shared__ float    sB1[64], sB2[64];
  __shared__ int      sPairs[496];
  __shared__ _Float16 sAct[8 * 16 * 64];  // per-wave D->A transpose slab

  const int tid = threadIdx.x;

  // edge_index passthrough: 496*256*32 == 2*E exactly
  {
    size_t i = (size_t)blockIdx.x * 256 + tid;
    #pragma unroll 1
    for (int r = 0; r < 32; ++r) {
      out[OUT_EI + i] = (float)edge_index[i];
      i += (size_t)496 * 256;
    }
  }
  // stage f16 weights + pair table
  for (int i = tid; i < 2048; i += 256) {
    const int k = i >> 6, o = i & 63;
    sWm[i] = (_Float16)((k < 16) ? (We1[k * 64 + o] + We1[(k + 16) * 64 + o])
                                 : 0.f);
  }
  for (int i = tid; i < 4096; i += 256) sW2[i] = (_Float16)We2[i];
  for (int i = tid; i < 1024; i += 256) {
    const int k = i >> 4, c = i & 15;
    sW3[i] = (_Float16)((c < 5) ? We3[k * 5 + c] : 0.f);
  }
  for (int i = tid; i < 64; i += 256) { sB1[i] = be1[i]; sB2[i] = be2[i]; }
  for (int q = tid; q < 496; q += 256) {
    int rem = q, s = 0;
    while (rem >= 31 - s) { rem -= 31 - s; ++s; }
    sPairs[q] = s | ((s + 1 + rem) << 8);
  }
  __syncthreads();

  const int lane = tid & 31, wid = tid >> 5;
  const int lo = lane & 15, hi = lane >> 4;

  // Resident B tiles (wave32 layout: elem e -> K=(e&7)+8*(2*(e>>3)+hi), N=lo)
  v16h B1r[4], B2r[2][4], B3r[2];
  #pragma unroll
  for (int j = 0; j < 4; ++j) { v16h b = {};
    #pragma unroll
    for (int e = 0; e < 16; ++e) b[e] = sWm[((e & 7) + 8 * (2 * (e >> 3) + hi)) * 64 + 16 * j + lo];
    B1r[j] = b; }
  #pragma unroll
  for (int c = 0; c < 2; ++c)
    #pragma unroll
    for (int j = 0; j < 4; ++j) { v16h b = {};
      #pragma unroll
      for (int e = 0; e < 16; ++e)
        b[e] = sW2[(32 * c + (e & 7) + 8 * (2 * (e >> 3) + hi)) * 64 + 16 * j + lo];
      B2r[c][j] = b; }
  #pragma unroll
  for (int c = 0; c < 2; ++c) { v16h b = {};
    #pragma unroll
    for (int e = 0; e < 16; ++e) b[e] = sW3[(32 * c + (e & 7) + 8 * (2 * (e >> 3) + hi)) * 16 + lo];
    B3r[c] = b; }

  float bb1[4], bb2[4];
  #pragma unroll
  for (int j = 0; j < 4; ++j) { bb1[j] = sB1[16 * j + lo]; bb2[j] = sB2[16 * j + lo]; }
  const float b3v = (lo < 5) ? be3[lo] : 0.f;
  _Float16* act = &sAct[wid * 1024];

  // 496 blocks * 8 waves * 16 iters == 63488 tiles == 2048 graphs * 31
  #pragma unroll 1
  for (int it = 0; it < 16; ++it) {
    const int t = blockIdx.x * 128 + it * 8 + wid;
    const int g = t / 31, ti = t - g * 31;
    // A1: row M=lo, features 8*hi..8*hi+7 (K 16..31 zero-padded)
    const int pp = sPairs[ti * 16 + lo];
    const float* xs = ws_x + ((size_t)(g * NPG + (pp & 255))) * 16 + 8 * hi;
    const float* xd = ws_x + ((size_t)(g * NPG + (pp >> 8)))  * 16 + 8 * hi;
    v16h A1 = {};
    #pragma unroll
    for (int i = 0; i < 8; ++i) A1[i] = (_Float16)(0.5f * (xs[i] + xd[i]));
    if (it < 15) {                       // global_prefetch_b8 next tile
      const int t2 = t + 8, g2 = t2 / 31, ti2 = t2 - g2 * 31;
      const int p2 = sPairs[ti2 * 16 + lo];
      __builtin_prefetch(ws_x + ((size_t)(g2 * NPG + (p2 & 255))) * 16, 0, 1);
    }
    // Layer 1: 16x16 @ 16x64  (4 WMMA)
    #pragma unroll
    for (int j = 0; j < 4; ++j) {
      v8f C = {};
      C = WMMA_F16(A1, B1r[j], C);
      #pragma unroll
      for (int v = 0; v < 8; ++v)
        act[(v + 8 * hi) * 64 + 16 * j + lo] =
            (_Float16)fmaxf(C[v] + bb1[j], 0.f);
    }
    __syncthreads();
    v16h A2a = {}, A2b = {};
    #pragma unroll
    for (int e = 0; e < 16; ++e) {
      const int kl = (e & 7) + 8 * (2 * (e >> 3) + hi);
      A2a[e] = act[lo * 64 + kl];
      A2b[e] = act[lo * 64 + 32 + kl];
    }
    __syncthreads();
    // Layer 2: 16x64 @ 64x64  (8 WMMA)
    #pragma unroll
    for (int j = 0; j < 4; ++j) {
      v8f C = {};
      C = WMMA_F16(A2a, B2r[0][j], C);
      C = WMMA_F16(A2b, B2r[1][j], C);
      #pragma unroll
      for (int v = 0; v < 8; ++v)
        act[(v + 8 * hi) * 64 + 16 * j + lo] =
            (_Float16)fmaxf(C[v] + bb2[j], 0.f);
    }
    __syncthreads();
    v16h A3a = {}, A3b = {};
    #pragma unroll
    for (int e = 0; e < 16; ++e) {
      const int kl = (e & 7) + 8 * (2 * (e >> 3) + hi);
      A3a[e] = act[lo * 64 + kl];
      A3b[e] = act[lo * 64 + 32 + kl];
    }
    __syncthreads();
    // Layer 3: 16x64 @ 64x16(pad)  (2 WMMA)
    v8f D = {};
    D = WMMA_F16(A3a, B3r[0], D);
    D = WMMA_F16(A3b, B3r[1], D);
    // attr is symmetric in (s,d): write forward + reverse edge slots
    if (lo < 5) {
      #pragma unroll
      for (int v = 0; v < 8; ++v) {
        const int row = v + 8 * hi;
        const int p2 = sPairs[ti * 16 + row];
        const int ss = p2 & 255, dd = p2 >> 8;      // ss < dd
        const float val = D[v] + b3v;
        out[OUT_ATTR + ((size_t)g * EPG + ss * 31 + dd - 1) * 5 + lo] = val;
        out[OUT_ATTR + ((size_t)g * EPG + dd * 31 + ss) * 5 + lo]     = val;
      }
    }
  }
}

// ============================ launcher =====================================
extern "C" void kernel_launch(void* const* d_in, const int* in_sizes, int n_in,
                              void* d_out, int out_size, void* d_ws,
                              size_t ws_size, hipStream_t stream) {
  (void)in_sizes; (void)n_in; (void)out_size; (void)ws_size;
  const float* lat   = (const float*)d_in[0];
  const int*   batch = (const int*)d_in[1];
  const int*   eidx  = (const int*)d_in[2];
  const float* noise = (const float*)d_in[3];
  float* ws   = (float*)d_ws;
  float* ws_e = ws;                 // [B,16]
  float* ws_x = ws + (size_t)BG*16; // [N,16]
  float* out  = (float*)d_out;

  k_graph<<<BG, 64, 0, stream>>>(lat,
      (const float*)d_in[4],  (const float*)d_in[5],
      (const float*)d_in[6],  (const float*)d_in[7],
      (const float*)d_in[8],  (const float*)d_in[9],
      (const float*)d_in[10], (const float*)d_in[11], (const float*)d_in[12],
      out + OUT_NNUM, ws_e);

  GnnArgs P;
  for (int l = 0; l < 4; ++l) {
    const int b = 13 + 6 * l;
    P.W1[l]  = (const float*)d_in[b];     P.b1[l] = (const float*)d_in[b + 1];
    P.g1[l]  = (const float*)d_in[b + 2]; P.bt1[l] = (const float*)d_in[b + 3];
    P.W2[l]  = (const float*)d_in[b + 4]; P.b2[l] = (const float*)d_in[b + 5];
  }
  for (int l = 0; l < 3; ++l) {
    P.ng[l] = (const float*)d_in[37 + 2 * l];
    P.nb[l] = (const float*)d_in[38 + 2 * l];
  }
  P.Wf1 = (const float*)d_in[43]; P.bf1 = (const float*)d_in[44];
  P.Wf2 = (const float*)d_in[45]; P.bf2 = (const float*)d_in[46];
  P.Wf3 = (const float*)d_in[47]; P.bf3 = (const float*)d_in[48];
  k_gnn<<<BG / 8, 256, 0, stream>>>(P, ws_e, noise, batch, ws_x, out);

  k_edge<<<496, 256, 0, stream>>>(ws_x, eidx,
      (const float*)d_in[49], (const float*)d_in[50],
      (const float*)d_in[51], (const float*)d_in[52],
      (const float*)d_in[53], (const float*)d_in[54],
      out);
}
